// AttentionKernelIntegral_84241488543950
// MI455X (gfx1250) — matrix operand order repaired and checked by reading, not compiled, not run on hardware
//
#include <hip/hip_runtime.h>
#include <hip/hip_bf16.h>
#include <math.h>

// ---------------------------------------------------------------------------
// AttentionKernelIntegral (linear attention) for gfx1250 / MI455X.
// GEMMs on V_WMMA_F32_16X16X32_F16 (f16 in, f32 accumulate).
// Epilogues (instance-norm, 2D rotary, bias) in fp32.
// B=4, N=8192, C=256, H=8, D=64.
// ---------------------------------------------------------------------------

typedef __attribute__((ext_vector_type(4)))  _Float16 v4h;
typedef __attribute__((ext_vector_type(8)))  _Float16 v8h;
typedef __attribute__((ext_vector_type(16))) _Float16 v16h;
typedef __attribute__((ext_vector_type(8)))  float    v8f;

#define BATCH   4
#define SEQ     8192
#define CDIM    256
#define HEADS   8
#define DDIM    64
#define BN      (BATCH * SEQ)      // 32768
#define HD      (HEADS * DDIM)     // 512
#define EPSV    1e-5f
#define INV_N   (1.0f / (float)SEQ)

__device__ __forceinline__ v8f wmma16(v16h a, v16h b, v8f c) {
    return __builtin_amdgcn_wmma_f32_16x16x32_f16(false, a, false, b, (short)0, c,
                                                  false, false);
}
// A fragment: p -> &A[row][kk + lh*8]; halves [0..7] at +0, [8..15] at +16 (K)
__device__ __forceinline__ v16h frag_a(const _Float16* p) {
    v8h lo = *(const v8h*)(p);
    v8h hi = *(const v8h*)(p + 16);
    return __builtin_shufflevector(lo, hi, 0,1,2,3,4,5,6,7,8,9,10,11,12,13,14,15);
}
// B fragment: p -> &BT[col][kk + lh*8]; halves [0..7] at +0, [8..15] at +16 (K)
__device__ __forceinline__ v16h frag_b(const _Float16* p) {
    v8h lo = *(const v8h*)(p);
    v8h hi = *(const v8h*)(p + 16);
    return __builtin_shufflevector(lo, hi, 0,1,2,3,4,5,6,7,8,9,10,11,12,13,14,15);
}

// ===========================================================================
// Kernel 1: q/k/v = u_x @ W^T (M=32768, K=256), one head (64 cols) per block.
// Fused: instance-norm (k,v) + 2D rotary (q,k); f16 outputs; k,v transposed.
// grid = (512, 24), block = 256 (8 waves: 4 row-tiles x 2 col-tiles of 16x32)
// ===========================================================================
__global__ void k1_qkv_proj(const float* __restrict__ ux,
                            const float* __restrict__ pos,
                            const float* __restrict__ Wq,
                            const float* __restrict__ Wk,
                            const float* __restrict__ Wv,
                            _Float16* __restrict__ qbuf,  // [BN][512]
                            _Float16* __restrict__ kT,    // [b][h][d][n]
                            _Float16* __restrict__ vT)    // [b][h][e][n]
{
    extern __shared__ float smemf[];
    _Float16* smem = (_Float16*)smemf;
    const int ST = CDIM + 8;                 // 264 halves: keeps b128 frags aligned
    _Float16* lA = smem;                     // 64 x 264
    _Float16* lB = smem + 64 * ST;           // 64 x 264

    const int tid  = threadIdx.x;
    const int row0 = blockIdx.x * 64;
    const int cg   = blockIdx.y;
    const int matId = cg >> 3;               // 0=q 1=k 2=v
    const int head  = cg & 7;
    const float* W = (matId == 0) ? Wq : (matId == 1) ? Wk : Wv;

    // stage A: 64 rows x 256 f32 -> f16 (4096 float4)
#pragma unroll
    for (int i = 0; i < 16; ++i) {
        int j = tid + 256 * i;
        int r = j >> 6, c4 = j & 63;
        float4 t = ((const float4*)(ux + (size_t)(row0 + r) * CDIM))[c4];
        v4h hv = { (_Float16)t.x, (_Float16)t.y, (_Float16)t.z, (_Float16)t.w };
        *(v4h*)&lA[r * ST + c4 * 4] = hv;
    }
    // stage B: 64 weight rows x 256, lB[col][k]
#pragma unroll
    for (int i = 0; i < 16; ++i) {
        int j = tid + 256 * i;
        int r = j >> 6, c4 = j & 63;
        float4 t = ((const float4*)(W + (size_t)(head * 64 + r) * CDIM))[c4];
        v4h hv = { (_Float16)t.x, (_Float16)t.y, (_Float16)t.z, (_Float16)t.w };
        *(v4h*)&lB[r * ST + c4 * 4] = hv;
    }
    __syncthreads();

    const int lane = tid & 31, w = tid >> 5;
    const int wr = w >> 1, wc = w & 1;       // 4x2 wave tiles (16 rows x 32 cols)
    const int lidx = lane & 15, lh = lane >> 4;

    v8f acc0 = {}, acc1 = {};
    const _Float16* pa  = &lA[(wr * 16 + lidx) * ST + lh * 8];
    const _Float16* pb0 = &lB[(wc * 32 + lidx) * ST + lh * 8];
    const _Float16* pb1 = pb0 + 16 * ST;
#pragma unroll
    for (int kk = 0; kk < CDIM; kk += 32) {
        v16h a  = frag_a(pa + kk);
        v16h b0 = frag_b(pb0 + kk);
        v16h b1 = frag_b(pb1 + kk);
        acc0 = wmma16(a, b0, acc0);
        acc1 = wmma16(a, b1, acc1);
    }
    __syncthreads();

    // stage 64x64 f32 result for fused norm + rope
    float* lR    = smemf;                    // 64 x 64 f32
    float* lMean = smemf + 64 * 64;
    float* lRstd = lMean + 64;
#pragma unroll
    for (int r = 0; r < 8; ++r) {
        lR[(wr * 16 + r + 8 * lh) * 64 + wc * 32 + lidx]      = acc0[r];
        lR[(wr * 16 + r + 8 * lh) * 64 + wc * 32 + 16 + lidx] = acc1[r];
    }
    __syncthreads();

    if (matId != 0 && tid < 64) {            // instance-norm stats over D=64
        float s = 0.f, s2 = 0.f;
        for (int c = 0; c < 64; ++c) { float x = lR[tid * 64 + c]; s += x; s2 += x * x; }
        float m = s * (1.f / 64.f);
        lMean[tid] = m;
        lRstd[tid] = rsqrtf(s2 * (1.f / 64.f) - m * m + EPSV);
    }
    __syncthreads();
    if (matId != 0) {                        // normalize in place
        int r = tid >> 2, c0 = (tid & 3) * 16;
        float m = lMean[r], rs = lRstd[r];
        for (int c = c0; c < c0 + 16; ++c)
            lR[r * 64 + c] = (lR[r * 64 + c] - m) * rs;
    }
    __syncthreads();

    if (matId == 0) {                        // q: rope, row-major [BN][512]
        int r = tid >> 2, c0 = (tid & 3) * 16;
        size_t grow = (size_t)row0 + r;
        int n = (int)(grow & (SEQ - 1));
        float tx = pos[n * 2 + 0] * 64.f;    // coord / MIN_FREQ
        float ty = pos[n * 2 + 1] * 64.f;
        for (int c = c0; c < c0 + 16; ++c) {
            int half = c >> 5, p = c & 31, j = p & 15;
            float freq = (half ? ty : tx) * powf(10000.f, -(float)j * (1.f / 16.f));
            float val = lR[r * 64 + c];
            float prt = (p < 16) ? -lR[r * 64 + c + 16] : lR[r * 64 + c - 16];
            qbuf[grow * HD + head * 64 + c] =
                (_Float16)(val * cosf(freq) + prt * sinf(freq));
        }
    } else if (matId == 1) {                 // k: rope, transposed [bh][d][n]
        int c = tid >> 2, r0 = (tid & 3) * 16;
        int half = c >> 5, p = c & 31, j = p & 15;
        float invf = powf(10000.f, -(float)j * (1.f / 16.f));
        int b = row0 >> 13;
        size_t base = (((size_t)(b * HEADS + head)) * 64 + c) * SEQ;
        for (int r = r0; r < r0 + 16; ++r) {
            int n = (row0 + r) & (SEQ - 1);
            float coord = half ? pos[n * 2 + 1] : pos[n * 2 + 0];
            float freq = coord * 64.f * invf;
            float val = lR[r * 64 + c];
            float prt = (p < 16) ? -lR[r * 64 + c + 16] : lR[r * 64 + c - 16];
            kT[base + n] = (_Float16)(val * cosf(freq) + prt * sinf(freq));
        }
    } else {                                 // v: normed, transposed [bh][e][n]
        int c = tid >> 2, r0 = (tid & 3) * 16;
        int b = row0 >> 13;
        size_t base = (((size_t)(b * HEADS + head)) * 64 + c) * SEQ;
        for (int r = r0; r < r0 + 16; ++r) {
            int n = (row0 + r) & (SEQ - 1);
            vT[base + n] = (_Float16)lR[r * 64 + c];
        }
    }
}

// ===========================================================================
// Kernel 2: partial dots = K_chunk^T @ V_chunk (64x64, K-chunk = 1024)
// grid = (32 bh, 8 chunks), block = 512 (16 waves, 4x4). f16 in, f32 out.
// ===========================================================================
__global__ void k2_dots_partial(const _Float16* __restrict__ kT,
                                const _Float16* __restrict__ vT,
                                float* __restrict__ partials) // [bh][8][64][64]
{
    const int bh = blockIdx.x, ch = blockIdx.y;
    const int tid = threadIdx.x, lane = tid & 31, w = tid >> 5;
    const int ti = w >> 2, tj = w & 3;
    const int lidx = lane & 15, lh = lane >> 4;
    const int d0 = ti * 16, e0 = tj * 16;
    const int n0 = ch * 1024;

    v8f acc = {};
    const _Float16* ka = kT + ((size_t)bh * 64 + d0 + lidx) * SEQ + n0 + lh * 8;
    const _Float16* vb = vT + ((size_t)bh * 64 + e0 + lidx) * SEQ + n0 + lh * 8;
#pragma unroll 8
    for (int kk = 0; kk < 1024; kk += 32) {
        v16h a = frag_a(ka + kk);
        v16h b = frag_b(vb + kk);
        acc = wmma16(a, b, acc);
    }
    float* pp = partials + (((size_t)bh * 8) + ch) * 4096;
#pragma unroll
    for (int r = 0; r < 8; ++r)
        pp[(d0 + r + 8 * lh) * 64 + e0 + lidx] = acc[r];
}

// Kernel 2b: deterministic chunk reduction (f32), fold in 1/N
__global__ void k2b_dots_reduce(const float* __restrict__ partials,
                                float* __restrict__ dots) // [bh][64][64]
{
    const int bh = blockIdx.x, tid = threadIdx.x;
    for (int i = 0; i < 16; ++i) {
        int idx = tid + 256 * i;
        float s = 0.f;
#pragma unroll
        for (int ch = 0; ch < 8; ++ch)
            s += partials[(((size_t)bh * 8) + ch) * 4096 + idx];
        dots[(size_t)bh * 4096 + idx] = s * INV_N;
    }
}

// ===========================================================================
// Kernel 3: u = q @ dots (M=32768 per head, K=64, N=64)
// grid = (1024, 8), block = 256 (8 waves, 2x4, tile 32x64). A direct global.
// ===========================================================================
__global__ void k3_q_dots(const _Float16* __restrict__ qbuf,
                          const float* __restrict__ dots,
                          _Float16* __restrict__ ubuf)   // [BN][512]
{
    extern __shared__ float smemf[];
    _Float16* lBT = (_Float16*)smemf;        // 64 x 72 (dots^T as f16: [e][d])
    const int BST = 72;

    const int tid  = threadIdx.x;
    const int row0 = blockIdx.x * 32;
    const int head = blockIdx.y;
    const int bh = (row0 >> 13) * HEADS + head;

#pragma unroll
    for (int i = 0; i < 16; ++i) {           // transpose+convert 64x64 dots
        int j = tid + 256 * i;
        int d = j >> 6, e = j & 63;
        lBT[e * BST + d] = (_Float16)dots[(size_t)bh * 4096 + j];
    }
    __syncthreads();

    const int lane = tid & 31, w = tid >> 5;
    const int wr = w >> 2, wc = w & 3;
    const int lidx = lane & 15, lh = lane >> 4;

    v8f acc = {};
    const _Float16* qa = qbuf + (size_t)(row0 + wr * 16 + lidx) * HD
                         + head * 64 + lh * 8;
    const _Float16* pb = &lBT[(wc * 16 + lidx) * BST + lh * 8];
#pragma unroll
    for (int kk = 0; kk < 64; kk += 32) {
        v16h a = frag_a(qa + kk);
        v16h b = frag_b(pb + kk);
        acc = wmma16(a, b, acc);
    }
#pragma unroll
    for (int r = 0; r < 8; ++r)
        ubuf[(size_t)(row0 + wr * 16 + r + 8 * lh) * HD + head * 64 + wc * 16 + lidx] =
            (_Float16)acc[r];
}

// ===========================================================================
// Kernel 4: out = u @ Wo^T + bo (M=32768, K=512, N=256), fp32 output
// grid = (1024, 4), block = 256 (8 waves, 2x4, tile 32x64)
// ===========================================================================
__global__ void k4_out_proj(const _Float16* __restrict__ ubuf,
                            const float* __restrict__ Wo,  // [256][512]
                            const float* __restrict__ bo,
                            float* __restrict__ out)       // [BN][256]
{
    extern __shared__ float smemf[];
    _Float16* smem = (_Float16*)smemf;
    const int ST = HD + 8;                   // 520
    _Float16* lA = smem;                     // 32 x 520
    _Float16* lB = smem + 32 * ST;           // 64 x 520

    const int tid   = threadIdx.x;
    const int row0  = blockIdx.x * 32;
    const int cbase = blockIdx.y * 64;

#pragma unroll
    for (int i = 0; i < 8; ++i) {            // A: 32x512 f16 (2048 v8h)
        int j = tid + 256 * i;
        int r = j >> 6, c8 = j & 63;
        *(v8h*)&lA[r * ST + c8 * 8] =
            *(const v8h*)(ubuf + (size_t)(row0 + r) * HD + c8 * 8);
    }
#pragma unroll
    for (int i = 0; i < 32; ++i) {           // B: 64x512 f32 -> f16, lB[col][k]
        int j = tid + 256 * i;
        int r = j >> 7, c4 = j & 127;
        float4 t = ((const float4*)(Wo + (size_t)(cbase + r) * HD))[c4];
        v4h hv = { (_Float16)t.x, (_Float16)t.y, (_Float16)t.z, (_Float16)t.w };
        *(v4h*)&lB[r * ST + c4 * 4] = hv;
    }
    __syncthreads();

    const int lane = tid & 31, w = tid >> 5;
    const int wr = w >> 2, wc = w & 3;
    const int lidx = lane & 15, lh = lane >> 4;

    v8f acc = {};
    const _Float16* pa = &lA[(wr * 16 + lidx) * ST + lh * 8];
    const _Float16* pb = &lB[(wc * 16 + lidx) * ST + lh * 8];
#pragma unroll
    for (int kk = 0; kk < HD; kk += 32) {
        v16h a = frag_a(pa + kk);
        v16h b = frag_b(pb + kk);
        acc = wmma16(a, b, acc);
    }
    float bias = bo[cbase + wc * 16 + lidx];
#pragma unroll
    for (int r = 0; r < 8; ++r)
        out[(size_t)(row0 + wr * 16 + r + 8 * lh) * CDIM + cbase + wc * 16 + lidx] =
            acc[r] + bias;
}

// ===========================================================================
extern "C" void kernel_launch(void* const* d_in, const int* in_sizes, int n_in,
                              void* d_out, int out_size, void* d_ws, size_t ws_size,
                              hipStream_t stream) {
    const float* ux  = (const float*)d_in[0];
    const float* pos = (const float*)d_in[1];
    const float* Wq  = (const float*)d_in[2];
    const float* Wk  = (const float*)d_in[3];
    const float* Wv  = (const float*)d_in[4];
    const float* Wo  = (const float*)d_in[5];
    const float* bo  = (const float*)d_in[6];
    float* out = (float*)d_out;

    _Float16* qbuf = (_Float16*)d_ws;                     // 16,777,216 h
    _Float16* kT   = qbuf + (size_t)BN * HD;              // 16,777,216 h
    _Float16* vT   = kT   + (size_t)BN * HD;              // 16,777,216 h
    float* partials = (float*)(vT + (size_t)BN * HD);     //  1,048,576 f
    float* dots     = partials + (size_t)32 * 8 * 4096;   //    131,072 f
    _Float16* ubuf  = kT;  // kT dead after k2; reuse for u

    const size_t sh1 = (size_t)(64 * 264 + 64 * 264) * 2; //  67,584 B
    const size_t sh3 = (size_t)(64 * 72) * 2;             //   9,216 B
    const size_t sh4 = (size_t)(32 * 520 + 64 * 520) * 2; //  99,840 B

    k1_qkv_proj<<<dim3(BN / 64, 24), 256, sh1, stream>>>(ux, pos, Wq, Wk, Wv,
                                                         qbuf, kT, vT);
    k2_dots_partial<<<dim3(32, 8), 512, 0, stream>>>(kT, vT, partials);
    k2b_dots_reduce<<<dim3(32), 256, 0, stream>>>(partials, dots);
    k3_q_dots<<<dim3(BN / 32, HEADS), 256, sh3, stream>>>(qbuf, dots, ubuf);
    k4_out_proj<<<dim3(BN / 32, 4), 256, sh4, stream>>>(ubuf, Wo, bo, out);
}
